// DynamicEdgeConv_87806311399699
// MI455X (gfx1250) — matrix operand order repaired and checked by reading, not compiled, not run on hardware
//
#include <hip/hip_runtime.h>

typedef __attribute__((ext_vector_type(16))) _Float16 v16h;
typedef __attribute__((ext_vector_type(8)))  float    v8f;

constexpr int B_   = 64;
constexpr int NPER = 2048;
constexpr int K_   = 16;
constexpr int DIN  = 64;
constexpr int DOUT = 64;
constexpr int NTOT = B_ * NPER;   // 131072

__device__ __forceinline__ unsigned umin_(unsigned a, unsigned b){ return a < b ? a : b; }
__device__ __forceinline__ unsigned umax_(unsigned a, unsigned b){ return a > b ? a : b; }

__device__ __forceinline__ v8f wmma_f16(v16h a, v16h b, v8f c){
  // D = A(16x32 f16) * B(32x16 f16) + C(16x16 f32)
  return __builtin_amdgcn_wmma_f32_16x16x32_f16(false, a, false, b, (short)0, c, false, false);
}

// CDNA5 async copy: 16 bytes per lane, global -> LDS, tracked by ASYNCcnt.
__device__ __forceinline__ void async_cp_b128(unsigned lds_off, const void* gaddr){
  asm volatile("global_load_async_to_lds_b128 %0, %1, off"
               :: "v"(lds_off), "v"(gaddr) : "memory");
}
__device__ __forceinline__ void wait_async0(){
  asm volatile("s_wait_asynccnt 0" ::: "memory");
}

// ---------------------------------------------------------------------------
// Kernel 0: fold weight into (W1 - W2) and W2, converted to f16, row-major [o][d]
// ---------------------------------------------------------------------------
__global__ __launch_bounds__(256) void prep_weights(
    const float* __restrict__ W, _Float16* __restrict__ w1m2, _Float16* __restrict__ w2)
{
  int i = blockIdx.x * blockDim.x + threadIdx.x;
  if (i < DOUT * DIN){
    int o = i >> 6, d = i & 63;
    float wa = W[o * (2*DIN) + d];
    float wb = W[o * (2*DIN) + DIN + d];
    w1m2[o * DIN + d] = (_Float16)(wa - wb);
    w2[o * DIN + d]   = (_Float16)wb;
  }
}

// ---------------------------------------------------------------------------
// Kernel 1: per 16-point tile (one wave each):
//   x16[n][d] (f16), sq[n] = ||x_n||^2,
//   a[n][o] = (W1-W2) x_n + bias,  c[n][o] = W2 x_n     (via WMMA)
// ---------------------------------------------------------------------------
__global__ __launch_bounds__(256) void prep_points(
    const float* __restrict__ x, const _Float16* __restrict__ w1m2,
    const _Float16* __restrict__ w2, const float* __restrict__ bias,
    _Float16* __restrict__ x16, float* __restrict__ sq,
    float* __restrict__ aout, float* __restrict__ cout)
{
  int lane = threadIdx.x & 31;
  int wv   = threadIdx.x >> 5;
  int n0   = (blockIdx.x * 8 + wv) * 16;

  // ---- f16 copy + squared norms (2 lanes per row) ----
  {
    int row = n0 + (lane >> 1);
    int cb  = (lane & 1) * 32;
    const float* xr = x + (size_t)row * DIN + cb;
    _Float16* xo = x16 + (size_t)row * DIN + cb;
    float s = 0.f;
    #pragma unroll
    for (int j = 0; j < 32; ++j){ float v = xr[j]; s += v * v; xo[j] = (_Float16)v; }
    s += __shfl_xor(s, 1, 32);
    if ((lane & 1) == 0) sq[row] = s;
  }

  // ---- A operand: rows n0..n0+15, contiguous K (k = 16h + s), matching B ----
  const float* xr = x + (size_t)(n0 + (lane & 15)) * DIN + ((lane < 16) ? 0 : 16);
  v16h A0, A1;
  #pragma unroll
  for (int j = 0; j < 8; ++j){
    A0[j]     = (_Float16)xr[j];
    A0[8 + j] = (_Float16)xr[8 + j];
    A1[j]     = (_Float16)xr[32 + j];
    A1[8 + j] = (_Float16)xr[40 + j];
  }

  int moff  = (lane < 16) ? 0 : 8;
  int ocol  = lane & 15;
  int kboff = (lane < 16) ? 0 : 16;
  #pragma unroll
  for (int ot = 0; ot < 4; ++ot){
    int o0 = ot * 16;
    const _Float16* w1p = w1m2 + (size_t)(o0 + ocol) * DIN + kboff;
    const _Float16* w2p = w2   + (size_t)(o0 + ocol) * DIN + kboff;
    v16h B0 = *(const v16h*)w1p;
    v16h B1 = *(const v16h*)(w1p + 32);
    v16h C0 = *(const v16h*)w2p;
    v16h C1 = *(const v16h*)(w2p + 32);
    v8f accA = {}; v8f accC = {};
    accA = wmma_f16(A0, B0, accA);
    accA = wmma_f16(A1, B1, accA);
    accC = wmma_f16(A0, C0, accC);
    accC = wmma_f16(A1, C1, accC);
    float bv = bias[o0 + ocol];
    #pragma unroll
    for (int r = 0; r < 8; ++r){
      size_t n = (size_t)(n0 + r + moff);
      aout[n * DOUT + o0 + ocol] = accA[r] + bv;
      cout[n * DOUT + o0 + ocol] = accC[r];
    }
  }
}

// ---------------------------------------------------------------------------
// Kernel 2: Gram tiles via WMMA with async-LDS double-buffered column chunks.
// Block = 16 rows of one batch, 16 waves. Chunk = 256 columns (32 KB f16);
// wave w computes tile w of each chunk; per-lane register top-16 with
// wave-vote early-out; LDS merge.
// ---------------------------------------------------------------------------
__global__ __launch_bounds__(512) void knn_kernel(
    const _Float16* __restrict__ x16, const float* __restrict__ sq,
    unsigned* __restrict__ knn)
{
  __shared__ __align__(64) _Float16 xcols[2][256 * DIN];  // 2 x 32 KB
  __shared__ __align__(64) unsigned shbuf[8192];          // 32 KB: stage, then merge

  int tid  = threadIdx.x;
  int lane = tid & 31;
  int wv   = tid >> 5;            // 0..15
  int b    = blockIdx.x >> 7;
  int n0   = (blockIdx.x & 127) * 16;
  size_t base = (size_t)b * NPER;
  const char* gcol = (const char*)(x16 + base * DIN);     // batch column base (128 B/row)

  unsigned lds0 = (unsigned)(size_t)(&xcols[0][0]);
  unsigned lds1 = (unsigned)(size_t)(&xcols[1][0]);
  unsigned coff = (unsigned)tid * 16u;

  // ---- kick off chunk 0 copy (32 KB: 4 x b128 per thread) ----
  #pragma unroll
  for (int k2 = 0; k2 < 4; ++k2)
    async_cp_b128(lds0 + coff + k2 * 8192u, gcol + coff + k2 * 8192u);

  // ---- A operand (rows n0..n0+15) + row norms, overlapped with the copy ----
  const _Float16* xrow = x16 + (base + n0 + (lane & 15)) * DIN + ((lane < 16) ? 0 : 16);
  v16h A0, A1;
  #pragma unroll
  for (int j = 0; j < 8; ++j){
    A0[j] = xrow[j];      A0[8 + j] = xrow[8 + j];
    A1[j] = xrow[32 + j]; A1[8 + j] = xrow[40 + j];
  }
  float sqA[8];
  #pragma unroll
  for (int r = 0; r < 8; ++r) sqA[r] = sq[base + n0 + r + ((lane < 16) ? 0 : 8)];

  unsigned list[16];
  #pragma unroll
  for (int j = 0; j < 16; ++j) list[j] = 0xFFFFFFFFu;

  int myrow  = lane & 15;
  int colsel = (lane < 16) ? 0 : 8;
  int moff   = (lane < 16) ? 0 : 8;

  wait_async0();
  __syncthreads();                 // chunk 0 visible to all waves

  for (int c = 0; c < 8; ++c){
    const _Float16* xl = (c & 1) ? xcols[1] : xcols[0];

    // prefetch chunk c+1 into the other buffer (its readers finished at the
    // end-of-iteration barrier of chunk c-1)
    if (c + 1 < 8){
      unsigned dst = (c & 1) ? lds0 : lds1;
      size_t gb = (size_t)(c + 1) * 32768u;
      #pragma unroll
      for (int k2 = 0; k2 < 4; ++k2)
        async_cp_b128(dst + coff + k2 * 8192u, gcol + gb + coff + k2 * 8192u);
    }

    // ---- compute tile: columns c*256 + wv*16 .. +15, B operand from LDS ----
    int c0 = c * 256 + wv * 16;
    const _Float16* xc = xl + (size_t)(wv * 16 + (lane & 15)) * DIN + ((lane < 16) ? 0 : 16);
    v16h B0 = *(const v16h*)xc;
    v16h B1 = *(const v16h*)(xc + 32);
    v8f acc = {};
    acc = wmma_f16(A0, B0, acc);
    acc = wmma_f16(A1, B1, acc);

    float sqB = sq[base + c0 + (lane & 15)];
    #pragma unroll
    for (int r = 0; r < 8; ++r){
      float d = fmaxf(sqA[r] + sqB - 2.0f * acc[r], 0.0f);
      unsigned key = (__float_as_uint(d) & 0xFFFFF800u) | (unsigned)(c0 + (lane & 15));
      shbuf[(unsigned)(wv * 16 + r + moff) * 17u + (unsigned)(lane & 15)] = key;
    }
    // same-wave LDS readback transposes C-layout to row-per-lane
    unsigned cands[8];
    #pragma unroll
    for (int j = 0; j < 8; ++j)
      cands[j] = shbuf[(unsigned)(wv * 16 + myrow) * 17u + (unsigned)(colsel + j)];

    #pragma unroll
    for (int j = 0; j < 8; ++j){
      unsigned cand = cands[j];
      // wave-uniform early out: after warm-up almost every candidate is worse
      // than every lane's current 16th-best -> single s_cbranch skips the sweep
      if (__any(cand < list[15])){
        #pragma unroll
        for (int q = 0; q < 16; ++q){    // branchless sorted insert
          unsigned lo = umin_(list[q], cand);
          unsigned hi = umax_(list[q], cand);
          list[q] = lo; cand = hi;
        }
      }
    }

    __syncthreads();                     // all readers done with buffer (c&1)
    if (c + 1 < 8){ wait_async0(); __syncthreads(); }   // chunk c+1 ready
  }

  // ---- merge: 32 sorted lists per row, 16 rounds of pop-min ----
  int listid = wv * 2 + (lane >> 4);
  #pragma unroll
  for (int j = 0; j < 16; ++j)
    shbuf[((unsigned)(myrow * 32 + listid)) * 16u + (unsigned)j] = list[j];
  __syncthreads();

  unsigned ml[16];
  #pragma unroll
  for (int j = 0; j < 16; ++j)
    ml[j] = shbuf[((unsigned)(wv * 32 + lane)) * 16u + (unsigned)j];

  unsigned sel = 0;
  #pragma unroll
  for (int k = 0; k < K_; ++k){
    unsigned m = ml[0];
    #pragma unroll
    for (int off = 16; off >= 1; off >>= 1)
      m = umin_(m, (unsigned)__shfl_xor((int)m, off, 32));
    if (lane == k) sel = m;
    bool won = (ml[0] == m);             // keys unique -> single winner
    #pragma unroll
    for (int j = 0; j < 15; ++j) ml[j] = won ? ml[j + 1] : ml[j];
    ml[15] = won ? 0xFFFFFFFFu : ml[15];
  }
  if (lane < K_){
    size_t n = base + n0 + wv;
    knn[n * K_ + lane] = sel & 2047u;
  }
}

// ---------------------------------------------------------------------------
// Kernel 3: out[n][o] = max_k relu(a[n][o] + c[knn[n][k]][o])
// ---------------------------------------------------------------------------
__global__ __launch_bounds__(256) void edge_max(
    const float* __restrict__ aout, const float* __restrict__ cout,
    const unsigned* __restrict__ knn, float* __restrict__ out)
{
  int p = blockIdx.x * 4 + (threadIdx.x >> 6);
  int o = threadIdx.x & 63;
  size_t base = (size_t)(p >> 11) * NPER;
  float av = aout[(size_t)p * DOUT + o];
  float best = 0.0f;   // relu output is >= 0
  #pragma unroll
  for (int k = 0; k < K_; ++k){
    unsigned j = knn[(size_t)p * K_ + k];
    float cv = cout[(base + j) * DOUT + o];
    best = fmaxf(best, fmaxf(av + cv, 0.0f));
  }
  out[(size_t)p * DOUT + o] = best;
}

// ---------------------------------------------------------------------------
extern "C" void kernel_launch(void* const* d_in, const int* in_sizes, int n_in,
                              void* d_out, int out_size, void* d_ws, size_t ws_size,
                              hipStream_t stream)
{
  const float* x      = (const float*)d_in[0];
  // d_in[1] = batch indices: implied by layout (n / NPER), unused
  const float* weight = (const float*)d_in[2];
  const float* bias   = (const float*)d_in[3];

  char* ws = (char*)d_ws;
  _Float16* x16  = (_Float16*)(ws);                       // 16,777,216 B
  float*    sq   = (float*)   (ws + 16777216);            //    524,288 B
  float*    aout = (float*)   (ws + 17301504);            // 33,554,432 B
  float*    cout_= (float*)   (ws + 50855936);            // 33,554,432 B
  _Float16* w1m2 = (_Float16*)(ws + 84410368);            //      8,192 B
  _Float16* w2   = (_Float16*)(ws + 84418560);            //      8,192 B
  unsigned* knn  = (unsigned*)(ws + 84426752);            //  8,388,608 B

  float* out = (float*)d_out;

  prep_weights<<<(DOUT * DIN + 255) / 256, 256, 0, stream>>>(weight, w1m2, w2);
  prep_points<<<NTOT / 128, 256, 0, stream>>>(x, w1m2, w2, bias, x16, sq, aout, cout_);
  knn_kernel<<<B_ * (NPER / 16), 512, 0, stream>>>(x16, sq, knn);
  edge_max<<<NTOT / 4, 256, 0, stream>>>(aout, cout_, knn, out);
}
